// Encoder_64922725646611
// MI455X (gfx1250) — matrix operand order repaired and checked by reading, not compiled, not run on hardware
//
#include <hip/hip_runtime.h>
#include <stdint.h>
#include <stddef.h>

// ---------------------------------------------------------------------------
// Types for CDNA5 WMMA (wave32)
// ---------------------------------------------------------------------------
typedef __attribute__((ext_vector_type(16))) __bf16 v16bf;
typedef __attribute__((ext_vector_type(2)))  __bf16 v2bf;
typedef __attribute__((ext_vector_type(8)))  float  v8f;

struct __align__(16) U4 { uint32_t x, y, z, w; };
union Frag { v16bf v; U4 q[2]; uint32_t w[8]; };

#define S_LEN 512
#define BATCH 128
#define HID   128
#define NLAY  3
#define LDU   392   // 384 + 8 pad (bank-conflict-free B-frag loads)
#define LDW   136   // 128 + 8 pad
#define NWG   12    // 3 layers * 4 j-tile groups (32 hidden units each)

// ---------------------------------------------------------------------------
// Scalar helpers (prefer gfx1250 hardware conversions / transcendentals)
// ---------------------------------------------------------------------------
#if __has_builtin(__builtin_amdgcn_cvt_pk_bf16_f32)
__device__ __forceinline__ uint32_t pack2(float lo, float hi) {
  v2bf r = __builtin_amdgcn_cvt_pk_bf16_f32(lo, hi);   // v_cvt_pk_bf16_f32
  return __builtin_bit_cast(uint32_t, r);
}
__device__ __forceinline__ uint16_t f2bf(float f) {
  return (uint16_t)(pack2(f, 0.f) & 0xFFFFu);
}
#else
// Native fptrunc: clang lowers float->__bf16 with the hw cvt when available.
__device__ __forceinline__ uint16_t f2bf(float f) {
  __bf16 b = (__bf16)f;
  return __builtin_bit_cast(uint16_t, b);
}
__device__ __forceinline__ uint32_t pack2(float lo, float hi) {
  v2bf r;
  r[0] = (__bf16)lo;
  r[1] = (__bf16)hi;
  return __builtin_bit_cast(uint32_t, r);
}
#endif

#if __has_builtin(__builtin_amdgcn_tanhf)
__device__ __forceinline__ float tanh_hw(float x) {
  return __builtin_amdgcn_tanhf(x);                    // v_tanh_f32
}
#else
__device__ __forceinline__ float tanh_hw(float x) { return tanhf(x); }
#endif

#if __has_builtin(__builtin_amdgcn_rcpf)
__device__ __forceinline__ float sigmoidf_(float x) {
  return __builtin_amdgcn_rcpf(1.f + __expf(-x));      // v_rcp_f32
}
#else
__device__ __forceinline__ float sigmoidf_(float x) { return 1.f / (1.f + __expf(-x)); }
#endif

__device__ __forceinline__ v8f zero8() {
  v8f z;
#pragma unroll
  for (int i = 0; i < 8; ++i) z[i] = 0.f;
  return z;
}

__device__ __forceinline__ v8f wmma_bf16(v16bf a, v16bf b, v8f c) {
  // D = A(16x32 bf16) * B(32x16 bf16) + C(16x16 f32)
  return __builtin_amdgcn_wmma_f32_16x16x32_bf16(false, a, false, b, (short)0, c,
                                                 false, false);
}

// A-fragment (16x32 bf16, row-major source, ISA 7.12.2 layout):
// lanes 0-15: row M=lane, K in {k0..k0+7, k0+16..k0+23}
// lanes 16-31: row M=lane-16, K in {k0+8..k0+15, k0+24..k0+31}
__device__ __forceinline__ v16bf load_afrag_bf16(const uint16_t* base, int lda,
                                                 int m_off, int k0) {
  const int lane = threadIdx.x & 31;
  const int m = lane & 15;
  const int ks = (lane >> 4) << 3;            // 0 or 8
  const uint16_t* p = base + (size_t)(m_off + m) * lda + k0 + ks;
  Frag f;
  f.q[0] = *reinterpret_cast<const U4*>(p);
  f.q[1] = *reinterpret_cast<const U4*>(p + 16);
  return f.v;
}

// Same layout, fp32 source with on-the-fly bf16 conversion.
__device__ __forceinline__ v16bf load_afrag_f32(const float* base, int lda,
                                                int m_off, int k0) {
  const int lane = threadIdx.x & 31;
  const int m = lane & 15;
  const int ks = (lane >> 4) << 3;
  const float* p = base + (size_t)(m_off + m) * lda + k0 + ks;
  Frag f;
#pragma unroll
  for (int i = 0; i < 4; ++i) {
    f.w[i]     = pack2(p[2 * i],      p[2 * i + 1]);
    f.w[4 + i] = pack2(p[16 + 2 * i], p[16 + 2 * i + 1]);
  }
  return f.v;
}

// B-fragment (32x16 bf16). B[k][n] comes from weight row n (W/U stored
// [gate_row][k] row-major, so a column of B = a contiguous weight row):
// lanes 0-15: col N=lane, K in {k0..k0+7, k0+16..k0+23} ; lanes 16-31: col N=lane-16, K in {k0+8..k0+15, k0+24..k0+31}
__device__ __forceinline__ v16bf load_bfrag(const uint16_t* base, int ldb,
                                            int n_off, int k0) {
  const int lane = threadIdx.x & 31;
  const int n = lane & 15;
  const int ks = (lane >> 4) << 3;            // 0 or 8
  const uint16_t* p = base + (size_t)(n_off + n) * ldb + k0 + ks;
  Frag f;
  f.q[0] = *reinterpret_cast<const U4*>(p);
  f.q[1] = *reinterpret_cast<const U4*>(p + 16);
  return f.v;
}

// Grid-wide barrier (generation counter in device memory).
__device__ __forceinline__ void grid_sync(unsigned* cnt, unsigned* gen) {
  __syncthreads();
  if (threadIdx.x == 0) {
    __threadfence();
    volatile unsigned* vgen = (volatile unsigned*)gen;
    const unsigned g = *vgen;
    if (atomicAdd(cnt, 1u) == NWG - 1) {
      *(volatile unsigned*)cnt = 0u;
      __threadfence();
      atomicAdd(gen, 1u);
    } else {
      while (*vgen == g) __builtin_amdgcn_s_sleep(1);
    }
    __threadfence();
  }
  __syncthreads();
}

// ---------------------------------------------------------------------------
// Utility kernels
// ---------------------------------------------------------------------------
__global__ void cvt_f32_bf16(const float* __restrict__ src,
                             uint16_t* __restrict__ dst, int n2) {
  int i = blockIdx.x * blockDim.x + threadIdx.x;   // one thread = 2 elements
  if (i < n2)
    *reinterpret_cast<uint32_t*>(dst + 2 * i) = pack2(src[2 * i], src[2 * i + 1]);
}

__global__ void zero_u32(uint32_t* __restrict__ p, int n) {
  int i = blockIdx.x * blockDim.x + threadIdx.x;
  if (i < n) p[i] = 0u;
}

// ---------------------------------------------------------------------------
// Input projection: xp = x @ lin_w^T + lin_b  -> bf16
// grid 512 x 256; each wave: 16 rows x 128 cols, K=128
// ---------------------------------------------------------------------------
__global__ void proj_wmma(const float* __restrict__ x,
                          const uint16_t* __restrict__ lw,
                          const float* __restrict__ lb,
                          uint16_t* __restrict__ xp) {
  const int wave = threadIdx.x >> 5;
  const int lane = threadIdx.x & 31;
  const int row0 = blockIdx.x * 128 + wave * 16;

  v8f acc[8];
#pragma unroll
  for (int c = 0; c < 8; ++c) acc[c] = zero8();

  for (int k = 0; k < HID; k += 32) {
    v16bf a = load_afrag_f32(x, HID, row0, k);
#pragma unroll
    for (int c = 0; c < 8; ++c) {
      v16bf b = load_bfrag(lw, HID, c * 16, k);
      acc[c] = wmma_bf16(a, b, acc[c]);
    }
  }
  const int n = lane & 15;
  const int rb = (lane >> 4) << 3;
#pragma unroll
  for (int c = 0; c < 8; ++c) {
    const float bias = lb[c * 16 + n];
#pragma unroll
    for (int r = 0; r < 8; ++r) {
      const int row = row0 + rb + r;
      xp[(size_t)row * HID + c * 16 + n] = f2bf(acc[c][r] + bias);
    }
  }
}

// ---------------------------------------------------------------------------
// Persistent recurrent kernel: 12 WGs x 256 threads, LDS-resident weights.
// WG wg owns layer = wg/4, hidden cols j in [ (wg%4)*32, +32 ), i.e. gate
// rows {g*128 + j} for g=0..3, packed in LDS as row (g*32 + jj).
// ---------------------------------------------------------------------------
__global__ void lstm_persistent(const uint16_t* __restrict__ xp_bf,  // [S*B, H]
                                const uint16_t* __restrict__ W_bf,   // [L,4H,H]
                                const uint16_t* __restrict__ U_bf,   // [L,4H,LH]
                                const float* __restrict__ G,         // [L,H]
                                float* __restrict__ h_out,           // [L,B,H]
                                float* __restrict__ c_out,           // [L,B,H]
                                uint16_t* __restrict__ h_bf,         // [L,B,H]
                                uint16_t* __restrict__ hx_bf,        // [B, L*H]
                                unsigned* __restrict__ bar) {
  extern __shared__ uint16_t lds[];
  uint16_t* Us = lds;                     // 128 x LDU
  uint16_t* Ws = lds + 128 * LDU;         // 128 x LDW

  const int wg = blockIdx.x;
  const int layer = wg >> 2;
  const int jbase = (wg & 3) * 32;
  const int wave = threadIdx.x >> 5;
  const int lane = threadIdx.x & 31;
  const int bt = wave;                    // batch tile (16 rows)
  unsigned* cnt = bar;
  unsigned* gen = bar + 1;

  // ---- Stage this WG's weight slice into LDS (bf16, padded rows) ----
  for (int idx = threadIdx.x; idx < 128 * 48; idx += blockDim.x) {
    const int row = idx / 48, ch = idx % 48;          // 48 x 8 elems = 384
    const int g = row >> 5, jj = row & 31;
    const int grow = g * 128 + jbase + jj;
    const uint16_t* src = U_bf + ((size_t)layer * 512 + grow) * 384 + ch * 8;
    *reinterpret_cast<U4*>(Us + row * LDU + ch * 8) =
        *reinterpret_cast<const U4*>(src);
  }
  for (int idx = threadIdx.x; idx < 128 * 16; idx += blockDim.x) {
    const int row = idx / 16, ch = idx % 16;          // 16 x 8 elems = 128
    const int g = row >> 5, jj = row & 31;
    const int grow = g * 128 + jbase + jj;
    const uint16_t* src = W_bf + ((size_t)layer * 512 + grow) * 128 + ch * 8;
    *reinterpret_cast<U4*>(Ws + row * LDW + ch * 8) =
        *reinterpret_cast<const U4*>(src);
  }
  __syncthreads();

  v8f acc[8];

  for (int t = 0; t < S_LEN; ++t) {
    // ---- Phase U: acc = hx @ Us^T (feedback term, all WGs in parallel) ----
#pragma unroll
    for (int c = 0; c < 8; ++c) acc[c] = zero8();
    if (layer == 0 && t + 1 < S_LEN)
      __builtin_prefetch(xp_bf + ((size_t)(t + 1) * BATCH + bt * 16) * HID, 0, 1);
    for (int k = 0; k < NLAY * HID; k += 32) {
      v16bf a = load_afrag_bf16(hx_bf, NLAY * HID, bt * 16, k);
#pragma unroll
      for (int c = 0; c < 8; ++c) {
        v16bf b = load_bfrag(Us, LDU, c * 16, k);
        acc[c] = wmma_bf16(a, b, acc[c]);
      }
    }

    // ---- Layer phases (sequential chain inside the step) ----
    for (int l = 0; l < NLAY; ++l) {
      if (l == layer) {
        const uint16_t* inp = (l == 0)
            ? xp_bf + (size_t)t * BATCH * HID
            : h_bf + (size_t)(l - 1) * BATCH * HID;
        for (int k = 0; k < HID; k += 32) {
          v16bf a = load_afrag_bf16(inp, HID, bt * 16, k);
#pragma unroll
          for (int c = 0; c < 8; ++c) {
            v16bf b = load_bfrag(Ws, LDW, c * 16, k);
            acc[c] = wmma_bf16(a, b, acc[c]);
          }
        }
        // elementwise LSTM cell on D-fragment layout
        const int n = lane & 15;
        const int rb = (lane >> 4) << 3;
#pragma unroll
        for (int s = 0; s < 2; ++s) {          // j sub-tile
#pragma unroll
          for (int r = 0; r < 8; ++r) {
            const int b = bt * 16 + rb + r;
            const int j = jbase + s * 16 + n;
            const float iv = sigmoidf_(acc[0 + s][r]);
            const float fv = sigmoidf_(acc[2 + s][r]);
            const float gv = tanh_hw(acc[4 + s][r]);
            const float ov = sigmoidf_(acc[6 + s][r]);
            const size_t o = (size_t)l * (BATCH * HID) + (size_t)b * HID + j;
            const float cy = fv * c_out[o] + iv * gv;
            const float hy = ov * tanh_hw(cy);
            c_out[o] = cy;
            h_out[o] = hy;
            h_bf[o] = f2bf(hy);
          }
        }
      }
      grid_sync(cnt, gen);
    }

    // ---- Phase H: hx(t+1) = sigmoid(h . G) * h, concat over layers ----
    if (t + 1 < S_LEN) {
      // 384 (layer,batch) pairs split: 32 per WG, 4 per wave
#pragma unroll
      for (int pp = 0; pp < 4; ++pp) {
        const int p = wg * 32 + wave * 4 + pp;
        const int l = p >> 7, b = p & 127;
        const float* hr = h_out + (size_t)l * (BATCH * HID) + (size_t)b * HID;
        float sum = 0.f;
        float hv[4];
#pragma unroll
        for (int q = 0; q < 4; ++q) {
          hv[q] = hr[lane + 32 * q];
          sum += hv[q] * G[l * HID + lane + 32 * q];
        }
#pragma unroll
        for (int off = 16; off > 0; off >>= 1) sum += __shfl_xor(sum, off, 32);
        const float gh = sigmoidf_(sum);
#pragma unroll
        for (int q = 0; q < 4; ++q)
          hx_bf[(size_t)b * (NLAY * HID) + l * HID + lane + 32 * q] =
              f2bf(gh * hv[q]);
      }
    }
    grid_sync(cnt, gen);
  }
}

// ---------------------------------------------------------------------------
// Launch
// ---------------------------------------------------------------------------
extern "C" void kernel_launch(void* const* d_in, const int* in_sizes, int n_in,
                              void* d_out, int out_size, void* d_ws, size_t ws_size,
                              hipStream_t stream) {
  (void)in_sizes; (void)n_in; (void)out_size; (void)ws_size;
  const float* x     = (const float*)d_in[0];  // [512,128,128]
  const float* lin_w = (const float*)d_in[1];  // [128,128]
  const float* lin_b = (const float*)d_in[2];  // [128]
  const float* W     = (const float*)d_in[3];  // [3,512,128]
  const float* U     = (const float*)d_in[4];  // [3,512,384]
  const float* G     = (const float*)d_in[5];  // [3,128,1]

  char* ws = (char*)d_ws;
  const size_t OFF_XP  = 0;                    // 512*128*128 bf16 = 16 MiB
  const size_t OFF_LW  = OFF_XP + (size_t)S_LEN * BATCH * HID * 2;
  const size_t OFF_W   = OFF_LW + (size_t)HID * HID * 2;
  const size_t OFF_U   = OFF_W  + (size_t)NLAY * 512 * HID * 2;
  const size_t OFF_HBF = OFF_U  + (size_t)NLAY * 512 * 384 * 2;
  const size_t OFF_HX  = OFF_HBF + (size_t)NLAY * BATCH * HID * 2;
  const size_t OFF_BAR = OFF_HX  + (size_t)BATCH * NLAY * HID * 2;

  uint16_t* xp_bf = (uint16_t*)(ws + OFF_XP);
  uint16_t* lw_bf = (uint16_t*)(ws + OFF_LW);
  uint16_t* W_bf  = (uint16_t*)(ws + OFF_W);
  uint16_t* U_bf  = (uint16_t*)(ws + OFF_U);
  uint16_t* h_bf  = (uint16_t*)(ws + OFF_HBF);
  uint16_t* hx_bf = (uint16_t*)(ws + OFF_HX);
  unsigned* bar   = (unsigned*)(ws + OFF_BAR);

  float* h_out = (float*)d_out;                       // [3,128,128]
  float* c_out = (float*)d_out + NLAY * BATCH * HID;  // [3,128,128]

  // bf16 weight conversions (2 elements per thread, packed cvt)
  {
    int n2 = (HID * HID) / 2;
    cvt_f32_bf16<<<(n2 + 255) / 256, 256, 0, stream>>>(lin_w, lw_bf, n2);
  }
  {
    int n2 = (NLAY * 512 * HID) / 2;
    cvt_f32_bf16<<<(n2 + 255) / 256, 256, 0, stream>>>(W, W_bf, n2);
  }
  {
    int n2 = (NLAY * 512 * 384) / 2;
    cvt_f32_bf16<<<(n2 + 255) / 256, 256, 0, stream>>>(U, U_bf, n2);
  }
  // zero h/c outputs + bf16 state + hx + barrier (contiguous region in ws)
  {
    int n = 2 * NLAY * BATCH * HID;                   // d_out dwords
    zero_u32<<<(n + 255) / 256, 256, 0, stream>>>((uint32_t*)d_out, n);
    int m = (int)((OFF_BAR + 256 - OFF_HBF) / 4);     // h_bf + hx_bf + bar
    zero_u32<<<(m + 255) / 256, 256, 0, stream>>>((uint32_t*)(ws + OFF_HBF), m);
  }
  // input projection (bf16 WMMA)
  proj_wmma<<<S_LEN * BATCH / 128, 256, 0, stream>>>(x, lw_bf, lin_b, xp_bf);

  // persistent recurrence: 12 WGs, ~132 KB dynamic LDS each
  const size_t lds_bytes = (size_t)(128 * LDU + 128 * LDW) * sizeof(uint16_t);
  lstm_persistent<<<NWG, 256, lds_bytes, stream>>>(xp_bf, W_bf, U_bf, G,
                                                   h_out, c_out, h_bf, hx_bf, bar);
}
